// FFJORD_28415503631106
// MI455X (gfx1250) — matrix operand order repaired and checked by reading, not compiled, not run on hardware
//
#include <hip/hip_runtime.h>

typedef _Float16 h16;
typedef __attribute__((ext_vector_type(16))) _Float16 v16h;
typedef __attribute__((ext_vector_type(8)))  _Float16 v8h;
typedef __attribute__((ext_vector_type(8)))  float    v8f;

#define BATCH  8192
#define HID    1024
#define FE     64
#define MT     64      // batch rows per workgroup
#define LDH    1032    // padded f16 stride for 1024-wide activation buffers
#define LDA0   72      // padded f16 stride for 64-wide buffers
#define LDX    65      // padded f32 stride for state buffers
#define NSTEPS 16

// gfx1250 hardware tanh (V_TANH_F32, single TRANS op) when available.
#if defined(__has_builtin)
#if __has_builtin(__builtin_amdgcn_tanhf)
#define FAST_TANH(x) __builtin_amdgcn_tanhf(x)
#endif
#endif
#ifndef FAST_TANH
#define FAST_TANH(x) tanhf(x)
#endif

__device__ __forceinline__ v16h cat8(v8h lo, v8h hi) {
  return __builtin_shufflevector(lo, hi, 0,1,2,3,4,5,6,7,8,9,10,11,12,13,14,15);
}

// A fragment (16x32 f16, M x K) from LDS row-major [m][k] with stride ld.
// Layout per CDNA5 ISA: lanes 0-15 hold M=lane, K={kk..kk+7, kk+16..kk+23};
// lanes 16-31 hold M=lane-16, K={kk+8..kk+15, kk+24..kk+31}.
__device__ __forceinline__ v16h load_a(const h16* A, int ld, int m0, int kk, int lane) {
  const int m  = m0 + (lane & 15);
  const int kb = kk + ((lane & 16) ? 8 : 0);
  const h16* p = A + m * ld + kb;
  v8h lo = *(const v8h*)(p);
  v8h hi = *(const v8h*)(p + 16);
  return cat8(lo, hi);
}

// B fragment (32x16 f16, K x N) from a "column-major" weight buffer Bcm[n][k]
// (k contiguous, stride ldk). Lanes 0-15: N=lane, K={kk..kk+7, kk+16..kk+23};
// lanes 16-31: N=lane-16, K={kk+8..kk+15, kk+24..kk+31} (matches A fragment).
__device__ __forceinline__ v16h load_b(const h16* __restrict__ B, int ldk, int n0, int kk, int lane) {
  const int n  = n0 + (lane & 15);
  const int kb = kk + ((lane & 16) ? 8 : 0);
  const h16* p = B + n * ldk + kb;
  v8h lo = *(const v8h*)(p);
  v8h hi = *(const v8h*)(p + 16);
  return cat8(lo, hi);
}

__device__ __forceinline__ v8f wmma32(v16h a, v16h b, v8f c) {
  return __builtin_amdgcn_wmma_f32_16x16x32_f16(false, a, false, b, (short)0, c, false, false);
}

// Wide GEMM: [MT x K] @ [K x 1024] -> 1024-wide f16 LDS buffer (stride LDH).
// mode 0: Out = tanh(acc + bias[n] + tmul*trow[n])
// mode 1: Out = acc * (1 - Out^2)   (tanh' VJP, in-place on Out)
__device__ __forceinline__
void gemm_wide(const h16* Alds, int lda, int K,
               const h16* __restrict__ Bg, h16* Olds,
               int mode, const float* __restrict__ bias,
               const float* __restrict__ trow, float tmul,
               int wave, int lane)
{
  for (int i = 0; i < 8; ++i) {
    const int n0 = (wave * 8 + i) * 16;
    __builtin_prefetch(Bg + (n0 + (lane & 15)) * K, 0, 1);
    v8f acc[4] = {};
    for (int kk = 0; kk < K; kk += 32) {
      v16h bf = load_b(Bg, K, n0, kk, lane);
#pragma unroll
      for (int mt = 0; mt < 4; ++mt)
        acc[mt] = wmma32(load_a(Alds, lda, mt * 16, kk, lane), bf, acc[mt]);
    }
    const int n = n0 + (lane & 15);
    float badd = 0.0f;
    if (mode == 0) badd = bias[n] + tmul * trow[n];
#pragma unroll
    for (int mt = 0; mt < 4; ++mt) {
#pragma unroll
      for (int j = 0; j < 8; ++j) {
        const int m = mt * 16 + j + ((lane & 16) ? 8 : 0);
        const float v = acc[mt][j];
        if (mode == 0) {
          Olds[m * LDH + n] = (h16)FAST_TANH(v + badd);
        } else {
          const float hh = (float)Olds[m * LDH + n];
          Olds[m * LDH + n] = (h16)(v * (1.0f - hh * hh));
        }
      }
    }
  }
}

// Narrow GEMM: [MT x 1024] @ [1024 x 64].
// mode 0: sdx[m][n] = acc + b3[n]
// mode 1: strace[m] += acc * eps[m][n]  (Hutchinson trace)
__device__ __forceinline__
void gemm_narrow(const h16* Alds, const h16* __restrict__ Bg,
                 int mode, const float* __restrict__ b3,
                 float* sdx, float* strace, const h16* geps,
                 int wave, int lane)
{
  for (int tt = wave; tt < 16; tt += 8) {
    const int mt = tt >> 2, nt = tt & 3;
    v8f acc = {};
    for (int kk = 0; kk < HID; kk += 32)
      acc = wmma32(load_a(Alds, LDH, mt * 16, kk, lane),
                   load_b(Bg, HID, nt * 16, kk, lane), acc);
    const int n = nt * 16 + (lane & 15);
    if (mode == 0) {
      const float bb = b3[n];
#pragma unroll
      for (int j = 0; j < 8; ++j) {
        const int m = mt * 16 + j + ((lane & 16) ? 8 : 0);
        sdx[m * LDX + n] = acc[j] + bb;
      }
    } else {
#pragma unroll
      for (int j = 0; j < 8; ++j) {
        const int m = mt * 16 + j + ((lane & 16) ? 8 : 0);
        const float e = (float)geps[m * LDA0 + n];
        atomicAdd(&strace[m], acc[j] * e);
      }
    }
  }
}

// Convert/transpose weights into f16 "B-ready" buffers.
__global__ void ffjord_prep(const float* __restrict__ W1, const float* __restrict__ W2,
                            const float* __restrict__ W3,
                            h16* __restrict__ W1a, h16* __restrict__ W1b,
                            h16* __restrict__ W2a, h16* __restrict__ W2b,
                            h16* __restrict__ W3a, h16* __restrict__ W3b)
{
  const int i = blockIdx.x * blockDim.x + threadIdx.x;
  if (i < HID * HID) {
    const float v = W2[i];
    const int r = i >> 10, c = i & 1023;
    W2b[i] = (h16)v;                 // B = W2^T path: Bcm[n][k] = W2[n][k]
    W2a[c * HID + r] = (h16)v;       // B = W2   path: Bcm[n][k] = W2[k][n]
  }
  if (i < FE * HID) {
    {
      const float v = W1[i];         // W1 rows 0..63 (feature rows): i = k*1024 + n
      const int k = i >> 10, n = i & 1023;
      W1a[n * FE + k] = (h16)v;      // fwd:  B = W1[0:64,:]
      W1b[i] = (h16)v;               // bwd:  B = W1[0:64,:]^T  -> Bcm[n][k]=W1[n][k]
    }
    {
      const float v = W3[i];         // W3: i = k*64 + n
      const int k = i >> 6, n = i & 63;
      W3b[i] = (h16)v;               // bwd:  B = W3^T -> Bcm[n][k]=W3[n][k]
      W3a[n * HID + k] = (h16)v;     // fwd:  B = W3
    }
  }
}

__global__ __launch_bounds__(256)
void ffjord_main(const float* __restrict__ x_in,
                 const float* __restrict__ b1, const float* __restrict__ b2,
                 const float* __restrict__ b3,
                 const float* __restrict__ W1,   // for t-row (row 64) + bias fold
                 const float* __restrict__ eps,
                 const h16* __restrict__ W1a, const h16* __restrict__ W1b,
                 const h16* __restrict__ W2a, const h16* __restrict__ W2b,
                 const h16* __restrict__ W3a, const h16* __restrict__ W3b,
                 float* __restrict__ outz, float* __restrict__ outl)
{
  __shared__ __attribute__((aligned(16))) h16 hbuf1[MT * LDH];  // h1 / g1
  __shared__ __attribute__((aligned(16))) h16 hbuf2[MT * LDH];  // h2 / g2
  __shared__ __attribute__((aligned(16))) h16 a0[MT * LDA0];    // f16 eval input
  __shared__ __attribute__((aligned(16))) h16 geps[MT * LDA0];  // f16 eps tile
  __shared__ float sx[MT * LDX];    // current state x
  __shared__ float sdx[MT * LDX];   // current stage f(x) (dx part)
  __shared__ float slog[MT];
  __shared__ float strace[MT];

  const int tid  = threadIdx.x;
  const int lane = tid & 31;
  const int wave = tid >> 5;
  const int row0 = blockIdx.x * MT;
  const float* trow = W1 + 64 * HID;  // t row of W1

  for (int i = tid; i < MT * FE; i += 256) {
    const int m = i >> 6, c = i & 63;
    sx[m * LDX + c]    = x_in[(size_t)(row0 + m) * FE + c];
    geps[m * LDA0 + c] = (h16)eps[(size_t)(row0 + m) * FE + c];
  }
  if (tid < MT) slog[tid] = 0.0f;
  __syncthreads();

  const float hstep = 1.0f / (float)NSTEPS;
  for (int step = 0; step < NSTEPS; ++step) {
    const float t0 = (float)step * hstep;
    for (int stage = 0; stage < 4; ++stage) {
      const float alpha = (stage == 0) ? 0.0f : (stage == 3 ? hstep : 0.5f * hstep);
      const float tcur  = t0 + alpha;

      // build eval input a0 = f16(sx + alpha * k_prev); zero trace accumulators
      for (int i = tid; i < MT * FE; i += 256) {
        const int m = i >> 6, c = i & 63;
        float v = sx[m * LDX + c];
        if (stage) v += alpha * sdx[m * LDX + c];
        a0[m * LDA0 + c] = (h16)v;
      }
      if (tid < MT) strace[tid] = 0.0f;
      __syncthreads();

      // forward: h1 = tanh(a0@W1 + b1 + t*W1[64])
      gemm_wide(a0, LDA0, FE, W1a, hbuf1, 0, b1, trow, tcur, wave, lane);
      __syncthreads();
      // h2 = tanh(h1@W2 + b2)
      gemm_wide(hbuf1, LDH, HID, W2a, hbuf2, 0, b2, trow, 0.0f, wave, lane);
      __syncthreads();
      // dx = h2@W3 + b3
      gemm_narrow(hbuf2, W3a, 0, b3, sdx, strace, geps, wave, lane);
      __syncthreads();
      // backward: g2 = (eps@W3^T) * (1-h2^2)   (in-place over hbuf2)
      gemm_wide(geps, LDA0, FE, W3b, hbuf2, 1, b3, trow, 0.0f, wave, lane);
      __syncthreads();
      // g1 = (g2@W2^T) * (1-h1^2)              (in-place over hbuf1)
      gemm_wide(hbuf2, LDH, HID, W2b, hbuf1, 1, b3, trow, 0.0f, wave, lane);
      __syncthreads();
      // vJ = g1@W1[0:64,:]^T ; trace += vJ*eps
      gemm_narrow(hbuf1, W1b, 1, b3, sdx, strace, geps, wave, lane);
      __syncthreads();

      // RK4 accumulation, streamed through d_out (same thread owns same elems)
      const float wgt  = (stage == 0 || stage == 3) ? 1.0f : 2.0f;
      const float coef = hstep * wgt * (1.0f / 6.0f);
      for (int i = tid; i < MT * FE; i += 256) {
        const int m = i >> 6, c = i & 63;
        const size_t go = (size_t)(row0 + m) * FE + c;
        const float k = sdx[m * LDX + c];
        if (stage == 0) outz[go] = sx[m * LDX + c] + coef * k;
        else            outz[go] += coef * k;
      }
      if (tid < MT) {
        const float k = -strace[tid];
        if (stage == 0) outl[row0 + tid] = slog[tid] + coef * k;
        else            outl[row0 + tid] += coef * k;
      }
      __syncthreads();
    }
    // commit step: y_{n+1} back into LDS state
    for (int i = tid; i < MT * FE; i += 256) {
      const int m = i >> 6, c = i & 63;
      sx[m * LDX + c] = outz[(size_t)(row0 + m) * FE + c];
    }
    if (tid < MT) slog[tid] = outl[row0 + tid];
    __syncthreads();
  }
  // final state already resides in outz / outl
}

extern "C" void kernel_launch(void* const* d_in, const int* in_sizes, int n_in,
                              void* d_out, int out_size, void* d_ws, size_t ws_size,
                              hipStream_t stream) {
  (void)in_sizes; (void)n_in; (void)out_size; (void)ws_size;
  const float* x   = (const float*)d_in[0];
  const float* W1  = (const float*)d_in[1];
  const float* b1  = (const float*)d_in[2];
  const float* W2  = (const float*)d_in[3];
  const float* b2  = (const float*)d_in[4];
  const float* W3  = (const float*)d_in[5];
  const float* b3  = (const float*)d_in[6];
  const float* eps = (const float*)d_in[7];

  h16* w   = (h16*)d_ws;
  h16* W1a = w;                        // [1024][64]
  h16* W1b = W1a + FE * HID;           // [64][1024]
  h16* W2a = W1b + FE * HID;           // [1024][1024] (W2^T)
  h16* W2b = W2a + HID * HID;          // [1024][1024] (W2)
  h16* W3a = W2b + HID * HID;          // [64][1024]
  h16* W3b = W3a + FE * HID;           // [1024][64]

  ffjord_prep<<<(HID * HID + 255) / 256, 256, 0, stream>>>(W1, W2, W3,
                                                           W1a, W1b, W2a, W2b, W3a, W3b);

  float* outz = (float*)d_out;
  float* outl = outz + (size_t)BATCH * FE;
  ffjord_main<<<BATCH / MT, 256, 0, stream>>>(x, b1, b2, b3, W1, eps,
                                              W1a, W1b, W2a, W2b, W3a, W3b,
                                              outz, outl);
}
